// RWKVAttention_29463475651340
// MI455X (gfx1250) — matrix operand order repaired and checked by reading, not compiled, not run on hardware
//
#include <hip/hip_runtime.h>

// ---- fixed problem shape (from the reference) ----
#define BB      4
#define SEQL    4096
#define DMODEL  1024
#define NHEADS  16
#define HEADD   64
#define MROWS   (BB * SEQL)   // 16384

typedef __bf16 bf16_t;
typedef __attribute__((ext_vector_type(16))) __bf16 v16bf;
typedef __attribute__((ext_vector_type(8)))  __bf16 v8bf;
typedef __attribute__((ext_vector_type(4)))  __bf16 v4bf;
typedef __attribute__((ext_vector_type(8)))  float  v8f;
typedef __attribute__((ext_vector_type(4)))  float  v4f;

// ------------------------------------------------------------------
// CDNA5 async global->LDS copy (ASYNCcnt-tracked), ISA 15.18.3 op 98.
// lds_off = workgroup-relative LDS byte address (= low 32 bits of the
// generic pointer, per ISA 10.2 aperture rules).
// ------------------------------------------------------------------
__device__ __forceinline__ void async_copy_b128(void* lds_ptr, const void* gptr)
{
    unsigned lds_off = (unsigned)(unsigned long long)lds_ptr;
    asm volatile("global_load_async_to_lds_b128 %0, %1, off"
                 :: "v"(lds_off), "v"(gptr)
                 : "memory");
}

__device__ __forceinline__ void wait_asynccnt0()
{
#if __has_builtin(__builtin_amdgcn_s_wait_asynccnt)
    __builtin_amdgcn_s_wait_asynccnt(0);
#else
    asm volatile("s_wait_asynccnt 0x0" ::: "memory");
#endif
}

// ------------------------------------------------------------------
// f32 -> bf16 conversion (used for the 4 weight matrices, 1M elems each)
// ------------------------------------------------------------------
__global__ __launch_bounds__(256) void cvt_f32_bf16(const float* __restrict__ src,
                                                    bf16_t* __restrict__ dst, int n)
{
    int i = (blockIdx.x * 256 + threadIdx.x) * 4;
    if (i + 3 < n) {
        v4f f = *(const v4f*)(src + i);
        v4bf b;
        b[0] = (bf16_t)f[0];
        b[1] = (bf16_t)f[1];
        b[2] = (bf16_t)f[2];
        b[3] = (bf16_t)f[3];
        *(v4bf*)(dst + i) = b;
    }
}

// ------------------------------------------------------------------
// C(MxN,f32) = A(MxK) @ W^T + bias, with W given bf16 row-major (N x K)
// so B[k][n] = Wbf[n][k].  128x128 tile per 256-thread block (8 waves),
// each wave computes a 32x64 strip via 2x4 V_WMMA_F32_16X16X32_BF16,
// K-chunk = 64 (two 32-wide WMMA sub-steps per barrier epoch).
// bf16 operands are staged with GLOBAL_LOAD_ASYNC_TO_LDS_B128.
// ------------------------------------------------------------------
template <bool A_BF16>
__global__ __launch_bounds__(256) void gemm128x128_wmma_bf16(
    const void* __restrict__ Aptr,      // M x K, f32 (A_BF16=false) or bf16 (true)
    const bf16_t* __restrict__ Wbf,     // N x K bf16 row-major
    const float* __restrict__ bias,     // N
    float* __restrict__ C,              // M x N f32
    int M, int N, int K)
{
    __shared__ __attribute__((aligned(128))) bf16_t As[128][64];
    __shared__ __attribute__((aligned(128))) bf16_t Bs[128][64];   // Bs[n][k]

    const int tid   = threadIdx.x;
    const int lane  = tid & 31;
    const int wid   = tid >> 5;
    const int waveM = wid >> 1;        // 0..3  -> 32-row strip
    const int waveN = wid & 1;         // 0..1  -> 64-col strip
    const int lrow  = lane & 15;
    const int lhi   = lane >> 4;       // 0 or 1

    const int mtile = blockIdx.y * 128;
    const int ntile = blockIdx.x * 128;

    // staging assignment: thread -> (row 0..127, half of 64 cols)
    const int srow  = tid >> 1;        // 0..127
    const int scol  = (tid & 1) << 5;  // 0 or 32

    v8f acc[2][4];
#pragma unroll
    for (int mi = 0; mi < 2; ++mi)
#pragma unroll
        for (int ni = 0; ni < 4; ++ni)
#pragma unroll
            for (int e = 0; e < 8; ++e)
                acc[mi][ni][e] = 0.0f;

    for (int kc = 0; kc < K; kc += 64) {
        // ---------- stage A tile (128 x 64) ----------
        if constexpr (A_BF16) {
            const bf16_t* Ab = (const bf16_t*)Aptr;
            const bf16_t* s = Ab + (size_t)(mtile + srow) * K + kc + scol;
#pragma unroll
            for (int j = 0; j < 4; ++j)
                async_copy_b128(&As[srow][scol + 8 * j], s + 8 * j);
            if (kc + 64 < K)
                __builtin_prefetch((const void*)(s + 64), 0, 1);
        } else {
            const float* Af = (const float*)Aptr;
            const float* s = Af + (size_t)(mtile + srow) * K + kc + scol;
#pragma unroll
            for (int j = 0; j < 4; ++j) {
                v4f f0 = ((const v4f*)s)[2 * j];
                v4f f1 = ((const v4f*)s)[2 * j + 1];
                v8bf b;
#pragma unroll
                for (int e = 0; e < 4; ++e) {
                    b[e]     = (bf16_t)f0[e];
                    b[e + 4] = (bf16_t)f1[e];
                }
                *(v8bf*)&As[srow][scol + 8 * j] = b;
            }
            if (kc + 64 < K)
                __builtin_prefetch((const void*)(s + 64), 0, 1);
        }
        // ---------- stage B tile: Bs[n][k] = Wbf[ntile+n][kc+k] (async) ----------
        {
            const bf16_t* s = Wbf + (size_t)(ntile + srow) * K + kc + scol;
#pragma unroll
            for (int j = 0; j < 4; ++j)
                async_copy_b128(&Bs[srow][scol + 8 * j], s + 8 * j);
            if (kc + 64 < K)
                __builtin_prefetch((const void*)(s + 64), 0, 1);
        }
        wait_asynccnt0();
        __syncthreads();

        // ---------- two 32-wide WMMA sub-steps ----------
#pragma unroll
        for (int ks = 0; ks < 2; ++ks) {
            const int kb = ks * 32;
            // A 16x32 bf16: lanes 0-15 row = lane, K[0:8]+K[16:24];
            //               lanes 16-31 row = lane-16, K[8:16]+K[24:32]
            v16bf afrag[2];
#pragma unroll
            for (int mi = 0; mi < 2; ++mi) {
                const int ar = waveM * 32 + mi * 16 + lrow;
                v8bf lo = *(const v8bf*)&As[ar][kb + lhi * 8];
                v8bf hi = *(const v8bf*)&As[ar][kb + lhi * 8 + 16];
                afrag[mi] = __builtin_shufflevector(lo, hi,
                    0,1,2,3,4,5,6,7,8,9,10,11,12,13,14,15);
            }
            // B 32x16 bf16: lanes 0-15 col = lane, K[0:8]+K[16:24]; lanes 16-31 col = lane-16, K[8:16]+K[24:32]
            v16bf bfrag[4];
#pragma unroll
            for (int ni = 0; ni < 4; ++ni) {
                const int bc = waveN * 64 + ni * 16 + lrow;
                v8bf lo = *(const v8bf*)&Bs[bc][kb + lhi * 8];
                v8bf hi = *(const v8bf*)&Bs[bc][kb + lhi * 8 + 16];
                bfrag[ni] = __builtin_shufflevector(lo, hi,
                    0,1,2,3,4,5,6,7,8,9,10,11,12,13,14,15);
            }

#pragma unroll
            for (int mi = 0; mi < 2; ++mi)
#pragma unroll
                for (int ni = 0; ni < 4; ++ni)
                    acc[mi][ni] = __builtin_amdgcn_wmma_f32_16x16x32_bf16(
                        /*neg_a=*/false, afrag[mi], /*neg_b=*/false, bfrag[ni],
                        /*c_mod=*/(short)0, acc[mi][ni],
                        /*reuse_a=*/false, /*reuse_b=*/false);
        }
        __syncthreads();
    }

    // ---------- epilogue: C/D layout -> VGPR r holds M = r + 8*(lane>=16) ----------
#pragma unroll
    for (int mi = 0; mi < 2; ++mi) {
#pragma unroll
        for (int ni = 0; ni < 4; ++ni) {
            const int col  = ntile + waveN * 64 + ni * 16 + lrow;
            const float bv = bias[col];
#pragma unroll
            for (int r = 0; r < 8; ++r) {
                const int row = mtile + waveM * 32 + mi * 16 + lhi * 8 + r;
                C[(size_t)row * N + col] = acc[mi][ni][r] + bv;
            }
        }
    }
}

// ------------------------------------------------------------------
// WKV recurrence + sigmoid gate, fused; one thread per (b, h*Dh+d) channel.
// num_t = df*num + e^k * v ; den_t = df*den + e^k ; wkv uses pre-update state.
// exp(bonus+kc) = exp(bonus)*exp(kc) -> 1 exp per step for the recurrence.
// Writes gated = sigmoid(r)*wkv as bf16 for the output-projection GEMM.
// ------------------------------------------------------------------
__global__ __launch_bounds__(256) void wkv_scan(
    const float* __restrict__ rproj, const float* __restrict__ kproj,
    const float* __restrict__ vproj, const float* __restrict__ decay,
    const float* __restrict__ bonus, bf16_t* __restrict__ gated)
{
    const int idx = blockIdx.x * 256 + threadIdx.x;   // 0 .. BB*DMODEL-1
    const int b   = idx >> 10;                        // / DMODEL
    const int c   = idx & (DMODEL - 1);               // h*HEADD + d

    const float df = __expf(-__expf(decay[c]));       // exp(-exp(decay))
    const float eb = __expf(bonus[c]);                // exp(bonus)

    float num = 0.0f, den = 0.0f;
    const size_t base = (size_t)b * SEQL * DMODEL + c;

    for (int t = 0; t < SEQL; ++t) {
        const size_t off = base + (size_t)t * DMODEL;
        if (t + 8 < SEQL) {
            __builtin_prefetch((const void*)(kproj + off + 8 * DMODEL), 0, 1);
            __builtin_prefetch((const void*)(vproj + off + 8 * DMODEL), 0, 1);
            __builtin_prefetch((const void*)(rproj + off + 8 * DMODEL), 0, 1);
        }
        const float kt = kproj[off];
        const float vt = vproj[off];
        const float rt = rproj[off];

        const float kcl = fminf(fmaxf(kt, -10.0f), 10.0f);
        const float ek  = __expf(kcl);
        const float bw  = eb * ek;

        const float wkv = (num + bw * vt) / (den + bw + 1e-6f);
        num = num * df + ek * vt;
        den = den * df + ek;

        const float g = wkv / (1.0f + __expf(-rt));   // sigmoid(r) * wkv
        gated[off] = (bf16_t)g;
    }
}

// ------------------------------------------------------------------
// Host launcher
// ------------------------------------------------------------------
extern "C" void kernel_launch(void* const* d_in, const int* in_sizes, int n_in,
                              void* d_out, int out_size, void* d_ws, size_t ws_size,
                              hipStream_t stream)
{
    (void)in_sizes; (void)n_in; (void)out_size; (void)ws_size;

    const float* q     = (const float*)d_in[0];
    const float* ksrc  = (const float*)d_in[1];
    const float* vsrc  = (const float*)d_in[2];
    const float* Wr    = (const float*)d_in[3];
    const float* br    = (const float*)d_in[4];
    const float* Wk    = (const float*)d_in[5];
    const float* bk    = (const float*)d_in[6];
    const float* Wv    = (const float*)d_in[7];
    const float* bv    = (const float*)d_in[8];
    const float* Wo    = (const float*)d_in[9];
    const float* bo    = (const float*)d_in[10];
    const float* decay = (const float*)d_in[11];
    const float* bonus = (const float*)d_in[12];
    float*       out   = (float*)d_out;

    // workspace layout
    char* ws = (char*)d_ws;
    size_t off = 0;
    float*  rproj = (float*)(ws + off); off += (size_t)MROWS * DMODEL * sizeof(float);
    float*  kproj = (float*)(ws + off); off += (size_t)MROWS * DMODEL * sizeof(float);
    float*  vproj = (float*)(ws + off); off += (size_t)MROWS * DMODEL * sizeof(float);
    bf16_t* gated = (bf16_t*)(ws + off); off += (size_t)MROWS * DMODEL * sizeof(bf16_t);
    bf16_t* Wr_bf = (bf16_t*)(ws + off); off += (size_t)DMODEL * DMODEL * sizeof(bf16_t);
    bf16_t* Wk_bf = (bf16_t*)(ws + off); off += (size_t)DMODEL * DMODEL * sizeof(bf16_t);
    bf16_t* Wv_bf = (bf16_t*)(ws + off); off += (size_t)DMODEL * DMODEL * sizeof(bf16_t);
    bf16_t* Wo_bf = (bf16_t*)(ws + off); off += (size_t)DMODEL * DMODEL * sizeof(bf16_t);

    const dim3 blk(256);

    // weights -> bf16
    const int nW = DMODEL * DMODEL;
    const dim3 cg((nW / 4 + 255) / 256);
    cvt_f32_bf16<<<cg, blk, 0, stream>>>(Wr, Wr_bf, nW);
    cvt_f32_bf16<<<cg, blk, 0, stream>>>(Wk, Wk_bf, nW);
    cvt_f32_bf16<<<cg, blk, 0, stream>>>(Wv, Wv_bf, nW);
    cvt_f32_bf16<<<cg, blk, 0, stream>>>(Wo, Wo_bf, nW);

    // projections (WMMA bf16, f32 accumulate)
    const dim3 gg(DMODEL / 128, MROWS / 128);   // (8, 128)
    gemm128x128_wmma_bf16<false><<<gg, blk, 0, stream>>>(q,    Wr_bf, br, rproj, MROWS, DMODEL, DMODEL);
    gemm128x128_wmma_bf16<false><<<gg, blk, 0, stream>>>(ksrc, Wk_bf, bk, kproj, MROWS, DMODEL, DMODEL);
    gemm128x128_wmma_bf16<false><<<gg, blk, 0, stream>>>(vsrc, Wv_bf, bv, vproj, MROWS, DMODEL, DMODEL);

    // fused WKV scan + sigmoid gate
    wkv_scan<<<dim3((BB * DMODEL) / 256), blk, 0, stream>>>(rproj, kproj, vproj, decay, bonus, gated);

    // output projection
    gemm128x128_wmma_bf16<true><<<gg, blk, 0, stream>>>(gated, Wo_bf, bo, out, MROWS, DMODEL, DMODEL);
}
